// NMRModel_71665824301653
// MI455X (gfx1250) — compile-verified, matrix operand order checked by reading, not run on hardware
//
#include <hip/hip_runtime.h>
#include <hip/hip_bf16.h>

// ---------------------------------------------------------------------------
// GATv2 GNN for MI455X (gfx1250).
//  - All K=256 GEMMs on the WMMA f16 path (f32 accumulate).
//  - Edge tensors (ea, ep: 480000x256, the dominant HBM traffic) stored f16.
//  - f16 A-tiles staged to LDS with GLOBAL_LOAD_ASYNC_TO_LDS_B128 (ASYNCcnt
//    path), vectorized b128 loads otherwise.
// ---------------------------------------------------------------------------

typedef __attribute__((ext_vector_type(16))) _Float16 v16h;
typedef __attribute__((ext_vector_type(8)))  float    v8f;
typedef __attribute__((ext_vector_type(4)))  int      v4i;

#define N_NODES 30000
#define N_EDGES 480000
#define HID     256
#define HEADS   8
#define DHEAD   32
#define NLAYERS 6
#define BN_EPS  1e-5f

#ifndef __has_builtin
#define __has_builtin(x) 0
#endif

#if __has_builtin(__builtin_amdgcn_global_load_async_to_lds_b128)
#define USE_ASYNC_LDS 1
#else
#define USE_ASYNC_LDS 0
#endif

typedef __attribute__((address_space(1))) v4i* gv4i_ptr;
typedef __attribute__((address_space(3))) v4i* lv4i_ptr;

static inline int cdiv(long long a, long long b) { return (int)((a + b - 1) / b); }

// K-offset inside a 16x32 f16 A fragment (ISA 7.12.2: 16-bit A-matrix 16x32).
// half = lane>>4. elems 0..7 -> K = half*8 + e ; elems 8..15 -> K = 16 + half*8 + (e-8)
__device__ __forceinline__ int kofs(int e, int hlf) {
    return (e < 8) ? (hlf * 8 + e) : (16 + hlf * 8 + (e - 8));
}

// Ordered-uint encoding for float atomic max.
__device__ __forceinline__ unsigned fkey(float f) {
    unsigned u = __float_as_uint(f);
    return (u & 0x80000000u) ? ~u : (u | 0x80000000u);
}
__device__ __forceinline__ float funkey(unsigned k) {
    unsigned u = (k & 0x80000000u) ? (k & 0x7FFFFFFFu) : ~k;
    return __uint_as_float(u);
}

__device__ __forceinline__ void wait_asynccnt0() {
#if __has_builtin(__builtin_amdgcn_s_wait_asynccnt)
    __builtin_amdgcn_s_wait_asynccnt(0);
#else
    asm volatile("s_wait_asynccnt 0x0" ::: "memory");
#endif
}

// ---------------------------------------------------------------------------
// Weight pack: fp32 [K=256][ncols] row-major -> f16 in B-fragment order:
// P[((kt*NT + nt)*32 + lane)*16 + e] = W[(kt*32 + kofs(e,half))*ncols + nt*16 + (lane&15)]
// so each lane's 16 B-fragment halfs are one contiguous 32-byte load.
// ---------------------------------------------------------------------------
__global__ void pack_w_f16(const float* __restrict__ W, _Float16* __restrict__ P, int ncols) {
    int nt_count = ncols >> 4;
    int total = 8 * nt_count * 32 * 16;
    int idx = blockIdx.x * blockDim.x + threadIdx.x;
    if (idx >= total) return;
    int e   = idx & 15;
    int l   = (idx >> 4) & 31;
    int nt  = (idx >> 9) % nt_count;
    int kt  = (idx >> 9) / nt_count;
    int hlf = l >> 4, r = l & 15;
    int k = kt * 32 + kofs(e, hlf);
    int n = nt * 16 + r;
    P[idx] = (_Float16)W[k * ncols + n];
}

// ---------------------------------------------------------------------------
// C[M x NCOLS] = act(A[M x 256] @ W + bias).  W pre-packed f16.
// NCG: number of 64-wide column groups (4 -> NCOLS=256, 2 -> NCOLS=128).
// AF16: A stored as f16 (enables async-to-LDS staging). CF16: C stored f16.
// Block = 256 threads = 8 waves; block tile = ROWS x NCOLS.
// A tile fully staged to LDS before any C write -> in-place C==A is safe.
// ---------------------------------------------------------------------------
template <int NCG, bool AF16, bool CF16>
__launch_bounds__(256)
__global__ void gemm_k256_wmma(const void* __restrict__ Av, const _Float16* __restrict__ P,
                               const float* __restrict__ bias, void* __restrict__ Cv,
                               int M, int act, float slope) {
    constexpr int ROWS  = (8 / NCG) * 16;   // 32 or 64
    constexpr int NCOLS = NCG * 64;         // 256 or 128
    constexpr int NT    = NCOLS / 16;       // 16 or 8

    __shared__ __align__(16) _Float16 As[ROWS][HID];

    const int tid      = threadIdx.x;
    const int blockRow = blockIdx.x * ROWS;

    // ---- stage A tile into LDS as f16 ----
    if (AF16) {
        const _Float16* A16 = (const _Float16*)Av;
        if (blockRow + ROWS <= M) {
            const _Float16* gsrc  = A16 + (size_t)blockRow * HID;  // tile is contiguous
            _Float16*       lbase = &As[0][0];
#if USE_ASYNC_LDS
            for (int c = tid * 8; c < ROWS * HID; c += 256 * 8) {
                __builtin_amdgcn_global_load_async_to_lds_b128(
                    (gv4i_ptr)(gsrc + c), (lv4i_ptr)(lbase + c), 0, 0);
            }
            wait_asynccnt0();
#else
            for (int c = tid * 8; c < ROWS * HID; c += 256 * 8)
                *(uint4*)(lbase + c) = *(const uint4*)(gsrc + c);
#endif
        } else {
            for (int i = tid; i < ROWS * HID; i += 256) {
                int rr = i >> 8, kk = i & 255;
                int gr = blockRow + rr;
                As[rr][kk] = (gr < M) ? A16[(size_t)gr * HID + kk] : (_Float16)0.f;
            }
        }
    } else {
        const float* A32 = (const float*)Av;
        for (int i = tid * 4; i < ROWS * HID; i += 256 * 4) {
            int rr = i >> 8, kk = i & 255;
            int gr = blockRow + rr;
            if (gr < M) {
                float4 v = *(const float4*)(A32 + (size_t)gr * HID + kk);
                As[rr][kk]     = (_Float16)v.x;
                As[rr][kk + 1] = (_Float16)v.y;
                As[rr][kk + 2] = (_Float16)v.z;
                As[rr][kk + 3] = (_Float16)v.w;
            } else {
                As[rr][kk] = As[rr][kk + 1] = As[rr][kk + 2] = As[rr][kk + 3] = (_Float16)0.f;
            }
        }
    }
    __syncthreads();

    const int wave = tid >> 5, lane = tid & 31;
    const int hlf  = lane >> 4, r = lane & 15;
    const int wr   = wave / NCG, wc = wave % NCG;
    const int rowBase = wr * 16;

    v8f acc0 = {}, acc1 = {}, acc2 = {}, acc3 = {};

#pragma unroll
    for (int kt = 0; kt < 8; ++kt) {
        union { v16h v; _Float16 h[16]; } a;
#pragma unroll
        for (int e = 0; e < 16; ++e)
            a.h[e] = As[rowBase + r][kt * 32 + kofs(e, hlf)];

        const _Float16* pb = P + (((size_t)kt * NT + wc * 4) * 32 + lane) * 16;
        v16h b0 = *(const v16h*)(pb);
        v16h b1 = *(const v16h*)(pb + 1 * 32 * 16);
        v16h b2 = *(const v16h*)(pb + 2 * 32 * 16);
        v16h b3 = *(const v16h*)(pb + 3 * 32 * 16);

        acc0 = __builtin_amdgcn_wmma_f32_16x16x32_f16(false, a.v, false, b0, (short)0, acc0, false, false);
        acc1 = __builtin_amdgcn_wmma_f32_16x16x32_f16(false, a.v, false, b1, (short)0, acc1, false, false);
        acc2 = __builtin_amdgcn_wmma_f32_16x16x32_f16(false, a.v, false, b2, (short)0, acc2, false, false);
        acc3 = __builtin_amdgcn_wmma_f32_16x16x32_f16(false, a.v, false, b3, (short)0, acc3, false, false);
    }

    // Store: C/D layout -> VGPR i: row = i + 8*half, col = lane&15 within 16-col tile.
#pragma unroll
    for (int t = 0; t < 4; ++t) {
        v8f acc = (t == 0) ? acc0 : (t == 1) ? acc1 : (t == 2) ? acc2 : acc3;
        int col = wc * 64 + t * 16 + r;
        float bval = bias ? bias[col] : 0.f;
#pragma unroll
        for (int i = 0; i < 8; ++i) {
            int gr = blockRow + rowBase + hlf * 8 + i;
            if (gr < M) {
                float v = acc[i] + bval;
                if (act) v = (v >= 0.f) ? v : slope * v;
                if (CF16) ((_Float16*)Cv)[(size_t)gr * NCOLS + col] = (_Float16)v;
                else      ((float*)Cv)[(size_t)gr * NCOLS + col] = v;
            }
        }
    }
}

// ---------------------------------------------------------------------------
// Small-K input projections (K=13 / K=10): plain FMA, trivial cost.
// ---------------------------------------------------------------------------
template <bool OUT16>
__global__ void proj_in(const float* __restrict__ X, const float* __restrict__ W,
                        const float* __restrict__ b, void* __restrict__ Cv, int M, int K) {
    size_t idx = (size_t)blockIdx.x * blockDim.x + threadIdx.x;
    if (idx >= (size_t)M * HID) return;
    int col = (int)(idx & 255);
    size_t row = idx >> 8;
    float s = b[col];
    for (int k = 0; k < K; ++k) s += X[row * K + k] * W[k * HID + col];
    if (OUT16) ((_Float16*)Cv)[idx] = (_Float16)s;
    else       ((float*)Cv)[idx] = s;
}

__global__ void fill_zero_u32(unsigned* __restrict__ p, size_t n) {
    size_t idx = (size_t)blockIdx.x * blockDim.x + threadIdx.x;
    if (idx < n) p[idx] = 0u;
}

// logits + segment-max (fused, per edge-head). ep is f16.
__global__ void edge_logits_k(const int* __restrict__ src, const int* __restrict__ dst,
                              const float* __restrict__ xl, const float* __restrict__ xr,
                              const _Float16* __restrict__ ep, const float* __restrict__ att,
                              float* __restrict__ logits, unsigned* __restrict__ maxb) {
    size_t idx = (size_t)blockIdx.x * blockDim.x + threadIdx.x;
    if (idx >= (size_t)N_EDGES * HEADS) return;
    int hh = (int)(idx & 7);
    size_t e = idx >> 3;
    int s = src[e], d = dst[e];
    const float*    pl = xl + (size_t)s * HID + hh * DHEAD;
    const float*    pr = xr + (size_t)d * HID + hh * DHEAD;
    const _Float16* pe = ep + e * HID + hh * DHEAD;
    const float*    pa = att + hh * DHEAD;
    float sum = 0.f;
#pragma unroll 8
    for (int c = 0; c < DHEAD; ++c) {
        float m = pl[c] + pr[c] + (float)pe[c];
        m = (m >= 0.f) ? m : 0.2f * m;
        sum += m * pa[c];
    }
    logits[idx] = sum;
    atomicMax(maxb + (size_t)d * HEADS + hh, fkey(sum));
}

// exp(logit - max) + segment-sum denominator (in-place on logits buffer)
__global__ void edge_exp_k(const int* __restrict__ dst, float* __restrict__ logits,
                           const unsigned* __restrict__ maxb, float* __restrict__ den) {
    size_t idx = (size_t)blockIdx.x * blockDim.x + threadIdx.x;
    if (idx >= (size_t)N_EDGES * HEADS) return;
    int hh = (int)(idx & 7);
    size_t e = idx >> 3;
    int d = dst[e];
    float mx = funkey(maxb[(size_t)d * HEADS + hh]);
    float ex = __expf(logits[idx] - mx);
    logits[idx] = ex;
    atomicAdd(den + (size_t)d * HEADS + hh, ex);
}

// alpha-weighted scatter-add of xl[src] into agg[dst]; 4 features per thread.
__global__ void edge_agg_k(const int* __restrict__ src, const int* __restrict__ dst,
                           const float* __restrict__ xl, const float* __restrict__ exv,
                           const float* __restrict__ den, float* __restrict__ agg) {
    size_t idx = (size_t)blockIdx.x * blockDim.x + threadIdx.x;
    if (idx >= (size_t)N_EDGES * (HID / 4)) return;
    int f = (int)(idx & 63) * 4;         // feature group
    size_t e = idx >> 6;
    int hh = f >> 5;
    int s = src[e], d = dst[e];
    float alpha = exv[e * HEADS + hh] / (den[(size_t)d * HEADS + hh] + 1e-16f);
    float4 v = *(const float4*)(xl + (size_t)s * HID + f);
    float* pa = agg + (size_t)d * HID + f;
    atomicAdd(pa + 0, v.x * alpha);
    atomicAdd(pa + 1, v.y * alpha);
    atomicAdd(pa + 2, v.z * alpha);
    atomicAdd(pa + 3, v.w * alpha);
}

// conv bias + BatchNorm(eval) + leaky(0.01) + residual, in-place on h
__global__ void node_update_k(const float* __restrict__ agg, const float* __restrict__ cb,
                              const float* __restrict__ g, const float* __restrict__ bta,
                              const float* __restrict__ rm, const float* __restrict__ rv,
                              float* __restrict__ h, int addres) {
    size_t idx = (size_t)blockIdx.x * blockDim.x + threadIdx.x;
    if (idx >= (size_t)N_NODES * HID) return;
    int f = (int)(idx & 255);
    float v = agg[idx] + cb[f];
    v = (v - rm[f]) * rsqrtf(rv[f] + BN_EPS) * g[f] + bta[f];
    v = (v >= 0.f) ? v : 0.01f * v;
    h[idx] = v + (addres ? h[idx] : 0.f);
}

// final [N,128] @ [128,1] + b
__global__ void out_head_k(const float* __restrict__ h2, const float* __restrict__ w2,
                           const float* __restrict__ b2, float* __restrict__ out) {
    int n = blockIdx.x * blockDim.x + threadIdx.x;
    if (n >= N_NODES) return;
    float s = b2[0];
    const float* p = h2 + (size_t)n * 128;
#pragma unroll 8
    for (int k = 0; k < 128; ++k) s += p[k] * w2[k];
    out[n] = s;
}

// ---------------------------------------------------------------------------
extern "C" void kernel_launch(void* const* d_in, const int* in_sizes, int n_in,
                              void* d_out, int out_size, void* d_ws, size_t ws_size,
                              hipStream_t stream) {
    (void)in_sizes; (void)n_in; (void)out_size; (void)ws_size;

    const float* x         = (const float*)d_in[0];
    const int*   eidx      = (const int*)d_in[1];
    const float* edge_attr = (const float*)d_in[2];
    const float* node_W    = (const float*)d_in[3];
    const float* node_b    = (const float*)d_in[4];
    const float* edge_W    = (const float*)d_in[5];
    const float* edge_b    = (const float*)d_in[6];
    const float* Wl        = (const float*)d_in[7];
    const float* bl        = (const float*)d_in[8];
    const float* Wr        = (const float*)d_in[9];
    const float* br        = (const float*)d_in[10];
    const float* We        = (const float*)d_in[11];
    const float* att       = (const float*)d_in[12];
    const float* conv_b    = (const float*)d_in[13];
    const float* Weu       = (const float*)d_in[14];
    const float* beu       = (const float*)d_in[15];
    const float* bn_g      = (const float*)d_in[16];
    const float* bn_b      = (const float*)d_in[17];
    const float* bn_rm     = (const float*)d_in[18];
    const float* bn_rv     = (const float*)d_in[19];
    const float* out_W1    = (const float*)d_in[20];
    const float* out_b1    = (const float*)d_in[21];
    const float* out_W2    = (const float*)d_in[22];
    const float* out_b2    = (const float*)d_in[23];

    const int* srcI = eidx;
    const int* dstI = eidx + N_EDGES;

    // ---- workspace carve-up (256B aligned) ----
    char* wp = (char*)d_ws;
    auto carve = [&](size_t bytes) -> char* {
        char* p = wp;
        wp += (bytes + 255) & ~(size_t)255;
        return p;
    };
    float*    h      = (float*)carve((size_t)N_NODES * HID * 4);
    _Float16* ea     = (_Float16*)carve((size_t)N_EDGES * HID * 2);   // f16 edge features
    float*    xl     = (float*)carve((size_t)N_NODES * HID * 4);
    float*    xr     = (float*)carve((size_t)N_NODES * HID * 4);
    _Float16* ep     = (_Float16*)carve((size_t)N_EDGES * HID * 2);   // f16 edge projections
    float*    logits = (float*)carve((size_t)N_EDGES * HEADS * 4);    // reused as exv
    unsigned* maxb   = (unsigned*)carve((size_t)N_NODES * HEADS * 4);
    float*    den    = (float*)carve((size_t)N_NODES * HEADS * 4);
    float*    agg    = (float*)carve((size_t)N_NODES * HID * 4);
    float*    h2     = (float*)carve((size_t)N_NODES * 128 * 4);
    _Float16* pWl    = (_Float16*)carve((size_t)NLAYERS * HID * HID * 2);
    _Float16* pWr    = (_Float16*)carve((size_t)NLAYERS * HID * HID * 2);
    _Float16* pWe    = (_Float16*)carve((size_t)NLAYERS * HID * HID * 2);
    _Float16* pWeu   = (_Float16*)carve((size_t)NLAYERS * HID * HID * 2);
    _Float16* pW1    = (_Float16*)carve((size_t)HID * 128 * 2);

    const int WMAT = HID * HID;                  // 65536
    const int packBlocks256 = cdiv(WMAT, 256);
    const int packBlocks128 = cdiv(HID * 128, 256);

    // ---- pack all weights to WMMA-fragment f16 layout ----
    for (int i = 0; i < NLAYERS; ++i) {
        pack_w_f16<<<packBlocks256, 256, 0, stream>>>(Wl  + (size_t)i * WMAT, pWl  + (size_t)i * WMAT, HID);
        pack_w_f16<<<packBlocks256, 256, 0, stream>>>(Wr  + (size_t)i * WMAT, pWr  + (size_t)i * WMAT, HID);
        pack_w_f16<<<packBlocks256, 256, 0, stream>>>(We  + (size_t)i * WMAT, pWe  + (size_t)i * WMAT, HID);
        pack_w_f16<<<packBlocks256, 256, 0, stream>>>(Weu + (size_t)i * WMAT, pWeu + (size_t)i * WMAT, HID);
    }
    pack_w_f16<<<packBlocks128, 256, 0, stream>>>(out_W1, pW1, 128);

    // ---- input projections ----
    proj_in<false><<<cdiv((size_t)N_NODES * HID, 256), 256, 0, stream>>>(x, node_W, node_b, h, N_NODES, 13);
    proj_in<true ><<<cdiv((size_t)N_EDGES * HID, 256), 256, 0, stream>>>(edge_attr, edge_W, edge_b, ea, N_EDGES, 10);

    const int gNode = cdiv(N_NODES, 32);
    const int gEdge = cdiv(N_EDGES, 32);
    const int gEH   = cdiv((size_t)N_EDGES * HEADS, 256);
    const int gEF4  = cdiv((size_t)N_EDGES * (HID / 4), 256);
    const int gNF   = cdiv((size_t)N_NODES * HID, 256);
    const int gNH   = cdiv((size_t)N_NODES * HEADS, 256);

    for (int i = 0; i < NLAYERS; ++i) {
        const size_t wOff = (size_t)i * WMAT;
        // xl = h @ Wl + bl ; xr = h @ Wr + br     (A f32, C f32)
        gemm_k256_wmma<4, false, false><<<gNode, 256, 0, stream>>>(h, pWl + wOff, bl + i * HID, xl, N_NODES, 0, 0.f);
        gemm_k256_wmma<4, false, false><<<gNode, 256, 0, stream>>>(h, pWr + wOff, br + i * HID, xr, N_NODES, 0, 0.f);
        // ep = ea @ We                             (A f16 async-staged, C f16)
        gemm_k256_wmma<4, true, true><<<gEdge, 256, 0, stream>>>(ea, pWe + wOff, nullptr, ep, N_EDGES, 0, 0.f);

        // reset segment buffers
        fill_zero_u32<<<gNH, 256, 0, stream>>>(maxb, (size_t)N_NODES * HEADS);
        fill_zero_u32<<<gNH, 256, 0, stream>>>((unsigned*)den, (size_t)N_NODES * HEADS);
        fill_zero_u32<<<gNF, 256, 0, stream>>>((unsigned*)agg, (size_t)N_NODES * HID);

        // attention softmax over incoming edges
        edge_logits_k<<<gEH, 256, 0, stream>>>(srcI, dstI, xl, xr, ep,
                                               att + (size_t)i * HEADS * DHEAD, logits, maxb);
        edge_exp_k<<<gEH, 256, 0, stream>>>(dstI, logits, maxb, den);
        edge_agg_k<<<gEF4, 256, 0, stream>>>(srcI, dstI, xl, logits, den, agg);

        // node update: conv bias + BN + leaky + residual (residual from layer 1 on)
        node_update_k<<<gNF, 256, 0, stream>>>(agg, conv_b + i * HID, bn_g + i * HID, bn_b + i * HID,
                                               bn_rm + i * HID, bn_rv + i * HID, h, (i >= 1) ? 1 : 0);

        // edge feature update, in-place f16 (A tile fully staged to LDS first)
        gemm_k256_wmma<4, true, true><<<gEdge, 256, 0, stream>>>(ea, pWeu + wOff, beu + i * HID, ea, N_EDGES, 0, 0.f);
    }

    // output MLP: leaky(h @ W1 + b1, 0.01) then @ W2 + b2
    gemm_k256_wmma<2, false, false><<<cdiv(N_NODES, 64), 256, 0, stream>>>(h, pW1, out_b1, h2, N_NODES, 1, 0.01f);
    out_head_k<<<cdiv(N_NODES, 256), 256, 0, stream>>>(h2, out_W2, out_b2, (float*)d_out);
}
